// SimpleGCNLayer_38611755991575
// MI455X (gfx1250) — matrix operand order repaired and checked by reading, not compiled, not run on hardware
//
#include <hip/hip_runtime.h>

#define Nn 8192
#define Ff 512

typedef __attribute__((ext_vector_type(16))) __bf16 bf16x16;
typedef __attribute__((ext_vector_type(8)))  __bf16 bf16x8;
typedef __attribute__((ext_vector_type(4)))  __bf16 bf16x4;
typedef __attribute__((ext_vector_type(8)))  float  f32x8;
typedef __attribute__((ext_vector_type(4)))  float  f32x4;

// Workgroup barrier that only drains LDS (DScnt) — keeps global B-fragment
// loads in flight across the barrier (CDNA5 split wait counters).
__device__ inline void lds_barrier() {
    asm volatile("s_wait_dscnt 0x0\n\t"
                 "s_barrier_signal -1\n\t"
                 "s_barrier_wait -1" ::: "memory");
}

// ---------------------------------------------------------------------------
// Kernel 1: s[i] = rsqrt( rowsum(adj[i,:]) + 1 )   (degree of A + I)
// ---------------------------------------------------------------------------
__global__ void gcn_rowsum(const float* __restrict__ adj, float* __restrict__ s) {
    const int row = blockIdx.x;
    const float* r = adj + (size_t)row * Nn;
    float p = 0.f;
    for (int c = threadIdx.x * 4; c < Nn; c += 256 * 4) {
        f32x4 v = *(const f32x4*)(r + c);
        p += v.x + v.y + v.z + v.w;
    }
    __shared__ float red[256];
    red[threadIdx.x] = p;
    __syncthreads();
    for (int off = 128; off > 0; off >>= 1) {
        if (threadIdx.x < off) red[threadIdx.x] += red[threadIdx.x + off];
        __syncthreads();
    }
    if (threadIdx.x == 0) s[row] = rsqrtf(red[0] + 1.0f);
}

// ---------------------------------------------------------------------------
// Kernel 2: Wt[n,k] = bf16(W[k,n])   (512x512, makes B-frags contiguous in K)
// ---------------------------------------------------------------------------
__global__ void gcn_transpose_w(const float* __restrict__ W, __bf16* __restrict__ Wt) {
    const int idx = blockIdx.x * 256 + threadIdx.x;   // 0 .. 262143
    const int n = idx >> 9, k = idx & 511;
    Wt[(size_t)n * Ff + k] = (__bf16)W[(size_t)k * Ff + n];
}

// ---------------------------------------------------------------------------
// Fragment helpers (CDNA5 16-bit WMMA layouts, wave32)
// ---------------------------------------------------------------------------
// A 16x32 bf16 frag from LDS tile (row stride 48 elems = 96B, 16B aligned):
// lane L: M = L%16 ; K = [base, base+16), base = L<16?0:16
__device__ inline bf16x16 load_a_frag(const __bf16* base, int lane) {
    const int m  = lane & 15;
    const int kb = (lane < 16) ? 0 : 16;
    const __bf16* p = base + m * 48 + kb;
    bf16x8 lo = *(const bf16x8*)(p);        // K = kb .. kb+7
    bf16x8 hi = *(const bf16x8*)(p + 8);    // K = kb+8 .. kb+15
    return __builtin_shufflevector(lo, hi, 0,1,2,3,4,5,6,7,8,9,10,11,12,13,14,15);
}

// B 32x16 bf16 frag direct from global (K-major storage Bt[n][k], ldb mult of 16):
// lane L: N = L%16 ; K = [k0, k0+16) for lanes 0-15, [k0+16, k0+32) for lanes 16-31
__device__ inline bf16x16 load_b_frag(const __bf16* __restrict__ Bt, int n0, int k0,
                                      int lane, int ldb) {
    const __bf16* p = Bt + (size_t)(n0 + (lane & 15)) * ldb + k0 + ((lane < 16) ? 0 : 16);
    return *(const bf16x16*)p;   // 32B -> two global_load_b128
}

// Stage a 4-element f32 strip into the LDS bf16 tile (optionally adding I on diag)
__device__ inline void stage4(__bf16* dst, f32x4 v, int grow, int gcol0,
                              int srow, int skk, bool adddiag) {
    float t0 = v.x, t1 = v.y, t2 = v.z, t3 = v.w;
    if (adddiag) {
        if (grow == gcol0 + 0) t0 += 1.0f;
        if (grow == gcol0 + 1) t1 += 1.0f;
        if (grow == gcol0 + 2) t2 += 1.0f;
        if (grow == gcol0 + 3) t3 += 1.0f;
    }
    bf16x4 b;
    b.x = (__bf16)t0; b.y = (__bf16)t1; b.z = (__bf16)t2; b.w = (__bf16)t3;
    *(bf16x4*)(dst + srow * 48 + skk) = b;   // 8B ds_store
}

// ---------------------------------------------------------------------------
// Kernel 3: Yt[n,j] = bf16( s[j] * (X @ W)[j,n] )   — WMMA bf16, K = 512
// Block: 32 rows x 512 cols, 8 waves each own 64 cols (4 col-tiles).
// B frags are software-pipelined one K-step ahead.
// ---------------------------------------------------------------------------
__global__ __launch_bounds__(256) void gcn_xw(
    const float* __restrict__ x, const __bf16* __restrict__ Wt,
    const float* __restrict__ s, __bf16* __restrict__ Yt) {

    __shared__ __bf16 Asm[2][32 * 48];
    const int tid = threadIdx.x, lane = tid & 31, wave = tid >> 5;
    const int m0 = blockIdx.x * 32;       // row block (j)
    const int nw = wave * 64;             // wave's column base
    const int srow = tid >> 3, skk = (tid & 7) * 4;
    const float* arow = x + (size_t)(m0 + srow) * Ff;

    f32x8 acc[2][4] = {};
    bf16x16 bb[2][4];

    // prologue: stage K-step 0 (A -> LDS) and preload B frags for K-step 0
    stage4(Asm[0], *(const f32x4*)(arow + skk), 0, 0, srow, skk, false);
    #pragma unroll
    for (int ct = 0; ct < 4; ++ct)
        bb[0][ct] = load_b_frag(Wt, nw + ct * 16, 0, lane, Ff);
    lds_barrier();

    const int NS = Ff / 32;   // 16
    #pragma unroll 2
    for (int ks = 0; ks < NS; ++ks) {
        const int cur = ks & 1, nxt = cur ^ 1;

        // issue next-step B loads first: they stay in flight through this
        // step's WMMAs and across the DScnt-only barrier
        if (ks + 1 < NS) {
            #pragma unroll
            for (int ct = 0; ct < 4; ++ct)
                bb[nxt][ct] = load_b_frag(Wt, nw + ct * 16, (ks + 1) * 32, lane, Ff);
        }
        f32x4 anxt = {};
        if (ks + 1 < NS) anxt = *(const f32x4*)(arow + (ks + 1) * 32 + skk);

        const __bf16* Abuf = Asm[cur];
        bf16x16 a0 = load_a_frag(Abuf,           lane);
        bf16x16 a1 = load_a_frag(Abuf + 16 * 48, lane);
        #pragma unroll
        for (int ct = 0; ct < 4; ++ct) {
            acc[0][ct] = __builtin_amdgcn_wmma_f32_16x16x32_bf16(
                false, a0, false, bb[cur][ct], (short)0, acc[0][ct], false, false);
            acc[1][ct] = __builtin_amdgcn_wmma_f32_16x16x32_bf16(
                false, a1, false, bb[cur][ct], (short)0, acc[1][ct], false, false);
        }
        lds_barrier();
        if (ks + 1 < NS) stage4(Asm[nxt], anxt, 0, 0, srow, skk, false);
        lds_barrier();
    }

    // epilogue: scale rows by s[j], store transposed bf16
    #pragma unroll
    for (int rt = 0; rt < 2; ++rt) {
        #pragma unroll
        for (int v = 0; v < 8; ++v) {
            const int j = m0 + rt * 16 + v + ((lane < 16) ? 0 : 8);
            const float sv = s[j];
            #pragma unroll
            for (int ct = 0; ct < 4; ++ct) {
                const int n = nw + ct * 16 + (lane & 15);
                Yt[(size_t)n * Nn + j] = (__bf16)(sv * acc[rt][ct][v]);
            }
        }
    }
}

// ---------------------------------------------------------------------------
// Kernel 4: out = relu( s[i] * ( (A + I) @ Yt^T )[i,:] )   — WMMA bf16, K = 8192
// Identity folded into the staged A tile; each block owns a 32-row stripe
// across all 512 columns so adj is streamed exactly once.
// ---------------------------------------------------------------------------
__global__ __launch_bounds__(256) void gcn_aggregate(
    const float* __restrict__ adj, const __bf16* __restrict__ Yt,
    const float* __restrict__ s, float* __restrict__ out) {

    __shared__ __bf16 Asm[2][32 * 48];
    const int tid = threadIdx.x, lane = tid & 31, wave = tid >> 5;
    const int m0 = blockIdx.x * 32;
    const int nw = wave * 64;
    const int srow = tid >> 3, skk = (tid & 7) * 4;
    const float* arow = adj + (size_t)(m0 + srow) * Nn;

    f32x8 acc[2][4] = {};
    bf16x16 bb[2][4];

    stage4(Asm[0], *(const f32x4*)(arow + skk), m0 + srow, skk, srow, skk, true);
    #pragma unroll
    for (int ct = 0; ct < 4; ++ct)
        bb[0][ct] = load_b_frag(Yt, nw + ct * 16, 0, lane, Nn);
    lds_barrier();

    const int NS = Nn / 32;   // 256
    #pragma unroll 2
    for (int ks = 0; ks < NS; ++ks) {
        const int cur = ks & 1, nxt = cur ^ 1;

        if (ks + 1 < NS) {
            #pragma unroll
            for (int ct = 0; ct < 4; ++ct)
                bb[nxt][ct] = load_b_frag(Yt, nw + ct * 16, (ks + 1) * 32, lane, Nn);
        }
        f32x4 anxt = {};
        if (ks + 1 < NS) anxt = *(const f32x4*)(arow + (ks + 1) * 32 + skk);

        const __bf16* Abuf = Asm[cur];
        bf16x16 a0 = load_a_frag(Abuf,           lane);
        bf16x16 a1 = load_a_frag(Abuf + 16 * 48, lane);
        #pragma unroll
        for (int ct = 0; ct < 4; ++ct) {
            acc[0][ct] = __builtin_amdgcn_wmma_f32_16x16x32_bf16(
                false, a0, false, bb[cur][ct], (short)0, acc[0][ct], false, false);
            acc[1][ct] = __builtin_amdgcn_wmma_f32_16x16x32_bf16(
                false, a1, false, bb[cur][ct], (short)0, acc[1][ct], false, false);
        }
        lds_barrier();
        if (ks + 1 < NS)
            stage4(Asm[nxt], anxt, m0 + srow, (ks + 1) * 32 + skk, srow, skk, true);
        lds_barrier();
    }

    // epilogue: out[i,n] = relu( s[i] * C[i,n] )
    #pragma unroll
    for (int rt = 0; rt < 2; ++rt) {
        #pragma unroll
        for (int v = 0; v < 8; ++v) {
            const int i = m0 + rt * 16 + v + ((lane < 16) ? 0 : 8);
            const float sv = s[i];
            #pragma unroll
            for (int ct = 0; ct < 4; ++ct) {
                const int n = nw + ct * 16 + (lane & 15);
                const float val = sv * acc[rt][ct][v];
                out[(size_t)i * Ff + n] = val > 0.f ? val : 0.f;
            }
        }
    }
}

// ---------------------------------------------------------------------------
extern "C" void kernel_launch(void* const* d_in, const int* in_sizes, int n_in,
                              void* d_out, int out_size, void* d_ws, size_t ws_size,
                              hipStream_t stream) {
    const float* x   = (const float*)d_in[0];   // [8192, 512]
    const float* adj = (const float*)d_in[1];   // [8192, 8192]
    const float* W   = (const float*)d_in[2];   // [512, 512]
    float* out = (float*)d_out;                 // [8192, 512]

    char* ws = (char*)d_ws;
    float*  s  = (float*)ws;                                   // 32 KB
    __bf16* Wt = (__bf16*)(ws + 32 * 1024);                    // 512 KB
    __bf16* Yt = (__bf16*)(ws + 32 * 1024 + 512 * 1024);       // 8 MB  [512][8192]

    gcn_rowsum    <<<Nn,               256, 0, stream>>>(adj, s);
    gcn_transpose_w<<<(Ff * Ff) / 256, 256, 0, stream>>>(W, Wt);
    gcn_xw        <<<Nn / 32,          256, 0, stream>>>(x, Wt, s, Yt);
    gcn_aggregate <<<Nn / 32,          256, 0, stream>>>(adj, Yt, s, out);
}
